// Attention_21887153341027
// MI455X (gfx1250) — compile-verified, matrix operand order checked by reading, not run on hardware
//
#include <hip/hip_runtime.h>

// ---------------------------------------------------------------------------
// MI455X (gfx1250, wave32) fused attention block:
//   qkv = x @ Wqkv + b ; rmsnorm(q,k) ; flash-attn ; out = attn @ Wproj + b
// bf16 WMMA (fp32 accumulate); softmax / rmsnorm fp32.
// - All LDS fragment gathers are 2x ds_load_b128 (A-style layout everywhere).
// - Double-buffered async global->LDS staging (ASYNCcnt) overlaps copies with
//   WMMA compute; s_wait_asynccnt only after compute, before the barrier.
// - GEMM wave tile 32x64 (1.5 ds_load_b128 per WMMA).
// ---------------------------------------------------------------------------

typedef __bf16 bf16;
typedef __attribute__((ext_vector_type(16))) __bf16 v16bf;
typedef __attribute__((ext_vector_type(8)))  __bf16 v8bf;
typedef __attribute__((ext_vector_type(8)))  float  v8f;
typedef unsigned int uint;

#define B_   4
#define N_   2048
#define C_   1024
#define H_   16
#define D_   64
#define M_   (B_ * N_)       // 8192
#define QKVN (3 * C_)        // 3072

#ifndef __has_builtin
#define __has_builtin(x) 0
#endif

#if __has_builtin(__builtin_amdgcn_global_load_async_to_lds_b128)
#define HAVE_ASYNC_LDS 1
#else
#define HAVE_ASYNC_LDS 0
#endif

#if HAVE_ASYNC_LDS
typedef int v4i __attribute__((vector_size(16)));
typedef __attribute__((address_space(1))) v4i* as1_v4i;
typedef __attribute__((address_space(3))) v4i* as3_v4i;
#define GASYNC_B128(gptr, lptr)                                                \
    __builtin_amdgcn_global_load_async_to_lds_b128(                            \
        (as1_v4i)(uintptr_t)(gptr), (as3_v4i)(uint)(uintptr_t)(lptr), 0, 0)
#if __has_builtin(__builtin_amdgcn_s_wait_asynccnt)
#define WAIT_ASYNC() __builtin_amdgcn_s_wait_asynccnt(0)
#else
#define WAIT_ASYNC() asm volatile("s_wait_asynccnt 0" ::: "memory")
#endif
#endif

__device__ __forceinline__ v8f wmma_bf16(v16bf a, v16bf b, v8f c) {
    return __builtin_amdgcn_wmma_f32_16x16x32_bf16(
        false, a, false, b, (short)0, c, false, false);
}

// A-style fragment (ISA 7.12.2 16-bit 16x32 layout): lane = half*16 + r.
// Lane data = two contiguous 16B runs: K = 8*half+[0..7], K = 16+8*half+[0..7].
__device__ __forceinline__ v16bf frag_a(const bf16* base, int ld, int lane) {
    const int half = lane >> 4;
    const int r    = lane & 15;
    const bf16* row = base + r * ld + 8 * half;
    const v8bf lo = *(const v8bf*)(row);
    const v8bf hi = *(const v8bf*)(row + 16);
    return __builtin_shufflevector(lo, hi, 0, 1, 2, 3, 4, 5, 6, 7,
                                           8, 9, 10, 11, 12, 13, 14, 15);
}

__device__ __forceinline__ v8f v8f_zero() {
    v8f z = {0.f, 0.f, 0.f, 0.f, 0.f, 0.f, 0.f, 0.f};
    return z;
}

// ---------------------------------------------------------------------------
// fp32 -> bf16 (grid-stride)
// ---------------------------------------------------------------------------
__global__ __launch_bounds__(256) void cvt_f32_bf16(const float* __restrict__ in,
                                                    bf16* __restrict__ out, int n) {
    int i = blockIdx.x * blockDim.x + threadIdx.x;
    const int stride = gridDim.x * blockDim.x;
    for (; i < n; i += stride) out[i] = (bf16)in[i];
}

// fp32 [R][Cc] -> bf16 [Cc][R] (LDS-tiled transpose; R,Cc multiples of 32)
__global__ __launch_bounds__(256) void cvt_transpose_f32_bf16(
    const float* __restrict__ in, bf16* __restrict__ out, int R, int Cc) {
    __shared__ float tile[32][33];
    const int tx = threadIdx.x & 31;
    const int ty = threadIdx.x >> 5;
    const int c0 = blockIdx.x * 32;
    const int r0 = blockIdx.y * 32;
#pragma unroll
    for (int i = 0; i < 4; ++i) {
        const int r = ty + i * 8;
        tile[r][tx] = in[(size_t)(r0 + r) * Cc + c0 + tx];
    }
    __syncthreads();
#pragma unroll
    for (int i = 0; i < 4; ++i) {
        const int c = ty + i * 8;
        out[(size_t)(c0 + c) * R + r0 + tx] = (bf16)tile[tx][c];
    }
}

// ---------------------------------------------------------------------------
// Shared GEMM core: block tile 128M x 128N, 8 waves (4M x 2N),
// wave tile 32x64, K-step 32, double-buffered LDS (A and B^T both [128][32]).
// Tiles staged via async global->LDS; stage(nxt) issued before compute(cur).
// ---------------------------------------------------------------------------
#define GEMM_STAGE(bufA, bufB, aBase, bBase, k0)                               \
    do {                                                                       \
        _Pragma("unroll")                                                      \
        for (int i_ = 0; i_ < 2; ++i_) {                                       \
            const int c_ = t * 2 + i_; /* 512 x 16B chunks, [128 rows][4] */   \
            const char* ga_ = (const char*)((aBase) + (size_t)(c_ >> 2) * C_ + (k0)) + (c_ & 3) * 16; \
            const char* gb_ = (const char*)((bBase) + (size_t)(c_ >> 2) * C_ + (k0)) + (c_ & 3) * 16; \
            STAGE16(ga_, (char*)(bufA) + c_ * 16);                             \
            STAGE16(gb_, (char*)(bufB) + c_ * 16);                             \
        }                                                                      \
    } while (0)

#if HAVE_ASYNC_LDS
#define STAGE16(g, l) GASYNC_B128(g, l)
#define STAGE_FENCE() WAIT_ASYNC()
#else
#define STAGE16(g, l) (*(v4i_fb*)(l) = *(const v4i_fb*)(g))
#define STAGE_FENCE() ((void)0)
typedef int v4i_fb __attribute__((vector_size(16)));
#endif

__device__ __forceinline__ void gemm_core_128x128(
    const bf16* __restrict__ aRows,   // block's A rows base: a + m0*C_
    const bf16* __restrict__ bRows,   // block's B^T rows base: wT + n0*C_
    bf16 (*lds_a)[128 * 32], bf16 (*lds_bT)[128 * 32],
    int t, int lane, int wm, int wn, v8f acc[2][4]) {
    GEMM_STAGE(lds_a[0], lds_bT[0], aRows, bRows, 0);
    STAGE_FENCE();
    __syncthreads();
    for (int k0 = 0; k0 < C_; k0 += 32) {
        const int cur = (k0 >> 5) & 1;
        if (k0 + 32 < C_)
            GEMM_STAGE(lds_a[cur ^ 1], lds_bT[cur ^ 1], aRows, bRows, k0 + 32);
        const v16bf af0 = frag_a(lds_a[cur] + (wm * 32) * 32, 32, lane);
        const v16bf af1 = frag_a(lds_a[cur] + (wm * 32 + 16) * 32, 32, lane);
#pragma unroll
        for (int tt = 0; tt < 4; ++tt) {
            const v16bf bfr = frag_a(lds_bT[cur] + (wn * 64 + tt * 16) * 32, 32, lane);
            acc[0][tt] = wmma_bf16(af0, bfr, acc[0][tt]);
            acc[1][tt] = wmma_bf16(af1, bfr, acc[1][tt]);
        }
        STAGE_FENCE();      // wait for next buffer only after compute
        __syncthreads();
    }
}

// ---------------------------------------------------------------------------
// QKV GEMM: xb[8192,1024] @ WqkvT[3072,1024]^T + bias.
// Epilogue scatters: q,k -> [BH][N][64] ; v -> [BH][64][N] (transposed).
// ---------------------------------------------------------------------------
__global__ __launch_bounds__(256) void qkv_gemm(const bf16* __restrict__ xb,
                                                const bf16* __restrict__ wT,
                                                const float* __restrict__ bias,
                                                bf16* __restrict__ qo,
                                                bf16* __restrict__ ko,
                                                bf16* __restrict__ vo) {
    __shared__ __align__(16) bf16 lds_a[2][128 * 32];
    __shared__ __align__(16) bf16 lds_bT[2][128 * 32];

    const int t = threadIdx.x, lane = t & 31, wave = t >> 5;
    const int wm = wave & 3, wn = wave >> 2;
    const int m0 = blockIdx.x * 128, n0 = blockIdx.y * 128;

    v8f acc[2][4];
#pragma unroll
    for (int s = 0; s < 2; ++s)
#pragma unroll
        for (int i = 0; i < 4; ++i) acc[s][i] = v8f_zero();

    gemm_core_128x128(xb + (size_t)m0 * C_, wT + (size_t)n0 * C_,
                      lds_a, lds_bT, t, lane, wm, wn, acc);

    const int half = lane >> 4, ln = lane & 15;
#pragma unroll
    for (int tt = 0; tt < 4; ++tt) {
        const int n_g   = n0 + wn * 64 + tt * 16 + ln;
        const int which = n_g >> 10;                 // 0:q 1:k 2:v
        const int rem   = n_g & (C_ - 1);
        const int h     = rem >> 6;
        const int d     = rem & (D_ - 1);
        const float bia = bias[n_g];
#pragma unroll
        for (int sub = 0; sub < 2; ++sub)
#pragma unroll
            for (int j = 0; j < 8; ++j) {
                const int m_g = m0 + wm * 32 + sub * 16 + j + 8 * half;
                const int b   = m_g >> 11;
                const int ns  = m_g & (N_ - 1);
                const bf16 val = (bf16)(acc[sub][tt][j] + bia);
                if (which == 2)  // V transposed: [BH][D][N]
                    vo[(((size_t)b * H_ + h) * D_ + d) * N_ + ns] = val;
                else {
                    bf16* dst = (which == 0) ? qo : ko;
                    dst[(((size_t)b * H_ + h) * N_ + ns) * D_ + d] = val;
                }
            }
    }
}

// ---------------------------------------------------------------------------
// Projection GEMM: ab[8192,1024] @ WprojT^T + bias -> fp32 out [8192,1024]
// ---------------------------------------------------------------------------
__global__ __launch_bounds__(256) void proj_gemm(const bf16* __restrict__ ab,
                                                 const bf16* __restrict__ wT,
                                                 const float* __restrict__ bias,
                                                 float* __restrict__ out) {
    __shared__ __align__(16) bf16 lds_a[2][128 * 32];
    __shared__ __align__(16) bf16 lds_bT[2][128 * 32];

    const int t = threadIdx.x, lane = t & 31, wave = t >> 5;
    const int wm = wave & 3, wn = wave >> 2;
    const int m0 = blockIdx.x * 128, n0 = blockIdx.y * 128;

    v8f acc[2][4];
#pragma unroll
    for (int s = 0; s < 2; ++s)
#pragma unroll
        for (int i = 0; i < 4; ++i) acc[s][i] = v8f_zero();

    gemm_core_128x128(ab + (size_t)m0 * C_, wT + (size_t)n0 * C_,
                      lds_a, lds_bT, t, lane, wm, wn, acc);

    const int half = lane >> 4, ln = lane & 15;
#pragma unroll
    for (int tt = 0; tt < 4; ++tt) {
        const int n_g = n0 + wn * 64 + tt * 16 + ln;
        const float bia = bias[n_g];
#pragma unroll
        for (int sub = 0; sub < 2; ++sub)
#pragma unroll
            for (int j = 0; j < 8; ++j) {
                const int m_g = m0 + wm * 32 + sub * 16 + j + 8 * half;
                out[(size_t)m_g * C_ + n_g] = acc[sub][tt][j] + bia;
            }
    }
}

// ---------------------------------------------------------------------------
// RMSNorm over D=64 per head-row, fp32 math, one wave per row
// ---------------------------------------------------------------------------
__global__ __launch_bounds__(256) void rmsnorm_rows(bf16* __restrict__ x,
                                                    const float* __restrict__ w,
                                                    int nrows) {
    const int wave = threadIdx.x >> 5;
    const int lane = threadIdx.x & 31;
    const int row  = blockIdx.x * 8 + wave;
    if (row >= nrows) return;
    bf16* p = x + (size_t)row * D_;
    const float a0 = (float)p[lane];
    const float a1 = (float)p[lane + 32];
    float s = a0 * a0 + a1 * a1;
#pragma unroll
    for (int m = 16; m >= 1; m >>= 1) s += __shfl_xor(s, m, 32);
    const float inv = rsqrtf(s * (1.0f / (float)D_) + 1e-6f);
    p[lane]      = (bf16)(a0 * inv * w[lane]);
    p[lane + 32] = (bf16)(a1 * inv * w[lane + 32]);
}

// ---------------------------------------------------------------------------
// Flash attention. q,k: [BH][N][64] ; v: [BH][64][N] (transposed).
// Block = (b*h, 128-query tile), 8 waves x 16 query rows; online softmax fp32.
// K/V tiles double-buffered via async global->LDS.
// ---------------------------------------------------------------------------
__global__ __launch_bounds__(256) void attn_fwd(const bf16* __restrict__ q,
                                                const bf16* __restrict__ k,
                                                const bf16* __restrict__ v,
                                                bf16* __restrict__ o) {
    __shared__ __align__(16) bf16 lds_q[128 * 64];      // [row][d]   16KB
    __shared__ __align__(16) bf16 lds_k[2][64 * 64];    // [key][d]  2x8KB
    __shared__ __align__(16) bf16 lds_vT[2][64 * 64];   // [d][key]  2x8KB
    __shared__ __align__(16) bf16 lds_p[8 * 16 * 64];   //           16KB

    const int bh   = blockIdx.x;
    const int qb   = blockIdx.y;
    const int t    = threadIdx.x;
    const int lane = t & 31;
    const int wave = t >> 5;
    const int half = lane >> 4, ln = lane & 15;

    const bf16* qbase = q + ((size_t)bh * N_ + qb * 128) * D_;
    const bf16* kbase = k + (size_t)bh * N_ * D_;
    const bf16* vbase = v + (size_t)bh * D_ * N_;    // rows = d, stride N_

    // stage K/V tile for key block kb into buffer buf (2 x 16B chunks each)
#define ATTN_STAGE_KV(buf, kb)                                                  \
    do {                                                                        \
        _Pragma("unroll")                                                       \
        for (int i_ = 0; i_ < 2; ++i_) {                                        \
            const int c_ = t * 2 + i_;                                          \
            STAGE16((const char*)(kbase + (size_t)(kb) * D_) + c_ * 16,         \
                    (char*)lds_k[buf] + c_ * 16);                               \
            STAGE16((const char*)(vbase + (size_t)(c_ >> 3) * N_ + (kb)) + (c_ & 7) * 16, \
                    (char*)lds_vT[buf] + c_ * 16);                              \
        }                                                                       \
    } while (0)

    {   // stage Q block: 1024 x 16B chunks, 4/thread, contiguous
#pragma unroll
        for (int i = 0; i < 4; ++i) {
            const int c = t * 4 + i;
            STAGE16((const char*)qbase + c * 16, (char*)lds_q + c * 16);
        }
        ATTN_STAGE_KV(0, 0);
        STAGE_FENCE();
    }
    __syncthreads();
    const v16bf qa0 = frag_a(lds_q + wave * 16 * 64, 64, lane);
    const v16bf qa1 = frag_a(lds_q + wave * 16 * 64 + 32, 64, lane);

    v8f ot[4];
    float mrow[8], lrow[8];
#pragma unroll
    for (int i = 0; i < 4; ++i) ot[i] = v8f_zero();
#pragma unroll
    for (int j = 0; j < 8; ++j) { mrow[j] = -INFINITY; lrow[j] = 0.f; }

    const float scale = 0.125f;  // D^-0.5

    for (int kb = 0; kb < N_; kb += 64) {
        const int cur = (kb >> 6) & 1;
        if (kb + 64 < N_) ATTN_STAGE_KV(cur ^ 1, kb + 64);

        // S = Q K^T (B^T gather folded into A-style frag on [key][d])
        v8f st[4];
#pragma unroll
        for (int tt = 0; tt < 4; ++tt) {
            v8f c = v8f_zero();
            c = wmma_bf16(qa0, frag_a(lds_k[cur] + (tt * 16) * 64, 64, lane), c);
            c = wmma_bf16(qa1, frag_a(lds_k[cur] + (tt * 16) * 64 + 32, 64, lane), c);
            st[tt] = c;
        }
#pragma unroll
        for (int tt = 0; tt < 4; ++tt)
#pragma unroll
            for (int j = 0; j < 8; ++j) st[tt][j] *= scale;

        // online softmax; row stats per VGPR index j (row = j + 8*half)
#pragma unroll
        for (int j = 0; j < 8; ++j) {
            float mx = fmaxf(fmaxf(st[0][j], st[1][j]), fmaxf(st[2][j], st[3][j]));
#pragma unroll
            for (int m = 8; m >= 1; m >>= 1) mx = fmaxf(mx, __shfl_xor(mx, m, 32));
            const float mnew = fmaxf(mrow[j], mx);
            const float corr = __expf(mrow[j] - mnew);
            mrow[j] = mnew;
            float rs = 0.f;
#pragma unroll
            for (int tt = 0; tt < 4; ++tt) {
                const float pv = __expf(st[tt][j] - mnew);
                st[tt][j] = pv;
                rs += pv;
            }
#pragma unroll
            for (int m = 8; m >= 1; m >>= 1) rs += __shfl_xor(rs, m, 32);
            lrow[j] = lrow[j] * corr + rs;
#pragma unroll
            for (int tt = 0; tt < 4; ++tt) ot[tt][j] *= corr;
        }

        // P (C layout) -> wave-private LDS -> A fragments
        bf16* pbuf = lds_p + wave * 16 * 64;
#pragma unroll
        for (int tt = 0; tt < 4; ++tt)
#pragma unroll
            for (int j = 0; j < 8; ++j)
                pbuf[(j + 8 * half) * 64 + tt * 16 + ln] = (bf16)st[tt][j];
        __syncthreads();

        const v16bf pa0 = frag_a(pbuf, 64, lane);
        const v16bf pa1 = frag_a(pbuf + 32, 64, lane);
#pragma unroll
        for (int tt = 0; tt < 4; ++tt) {   // B from vT[d][key]: A-style gather
            ot[tt] = wmma_bf16(pa0, frag_a(lds_vT[cur] + (tt * 16) * 64, 64, lane), ot[tt]);
            ot[tt] = wmma_bf16(pa1, frag_a(lds_vT[cur] + (tt * 16) * 64 + 32, 64, lane), ot[tt]);
        }

        STAGE_FENCE();      // next K/V buffer ready only after compute
        __syncthreads();
    }

    // normalize, scatter to [B][N][C] bf16
    const int b = bh >> 4, h = bh & 15;
#pragma unroll
    for (int tt = 0; tt < 4; ++tt) {
        const int d = tt * 16 + ln;
#pragma unroll
        for (int j = 0; j < 8; ++j) {
            const int ns = qb * 128 + wave * 16 + j + 8 * half;
            o[((size_t)b * N_ + ns) * C_ + h * D_ + d] = (bf16)(ot[tt][j] / lrow[j]);
        }
    }
#undef ATTN_STAGE_KV
}

// ---------------------------------------------------------------------------
// Launch. Inputs (fp32): x, W_qkv, b_qkv, W_proj, b_proj, qn_w, kn_w
// Workspace (~75.5 MB): xb(=attn_out alias) | WqkvT | WprojT | q | k | vT
// ---------------------------------------------------------------------------
extern "C" void kernel_launch(void* const* d_in, const int* in_sizes, int n_in,
                              void* d_out, int out_size, void* d_ws, size_t ws_size,
                              hipStream_t stream) {
    const float* x     = (const float*)d_in[0];
    const float* Wqkv  = (const float*)d_in[1];
    const float* bqkv  = (const float*)d_in[2];
    const float* Wproj = (const float*)d_in[3];
    const float* bproj = (const float*)d_in[4];
    const float* qnw   = (const float*)d_in[5];
    const float* knw   = (const float*)d_in[6];

    char* ws = (char*)d_ws;
    size_t off = 0;
    auto alloc = [&](size_t bytes) -> void* {
        void* p = ws + off;
        off += (bytes + 255) & ~(size_t)255;
        return p;
    };
    bf16* xb   = (bf16*)alloc((size_t)M_ * C_ * 2);     // reused as attn_out
    bf16* wqbT = (bf16*)alloc((size_t)QKVN * C_ * 2);   // [3072][1024]
    bf16* wpbT = (bf16*)alloc((size_t)C_ * C_ * 2);     // [1024][1024]
    bf16* qbf  = (bf16*)alloc((size_t)B_ * H_ * N_ * D_ * 2);
    bf16* kbf  = (bf16*)alloc((size_t)B_ * H_ * N_ * D_ * 2);
    bf16* vbfT = (bf16*)alloc((size_t)B_ * H_ * D_ * N_ * 2);  // [BH][D][N]

    cvt_f32_bf16<<<4096, 256, 0, stream>>>(x, xb, M_ * C_);
    cvt_transpose_f32_bf16<<<dim3(QKVN / 32, C_ / 32), 256, 0, stream>>>(Wqkv, wqbT, C_, QKVN);
    cvt_transpose_f32_bf16<<<dim3(C_ / 32, C_ / 32), 256, 0, stream>>>(Wproj, wpbT, C_, C_);

    qkv_gemm<<<dim3(M_ / 128, QKVN / 128), 256, 0, stream>>>(xb, wqbT, bqkv, qbf, kbf, vbfT);

    const int nrows = B_ * H_ * N_;  // 131072
    rmsnorm_rows<<<nrows / 8, 256, 0, stream>>>(qbf, qnw, nrows);
    rmsnorm_rows<<<nrows / 8, 256, 0, stream>>>(kbf, knw, nrows);

    attn_fwd<<<dim3(B_ * H_, N_ / 128), 256, 0, stream>>>(qbf, kbf, vbfT, xb /*attn_out*/);

    proj_gemm<<<dim3(M_ / 128, C_ / 128), 256, 0, stream>>>(xb, wpbT, bproj, (float*)d_out);
}
